// DCRNN_46995532153058
// MI455X (gfx1250) — matrix operand order repaired and checked by reading, not compiled, not run on hardware
//
#include <hip/hip_runtime.h>
#include <hip/hip_bf16.h>

// ---------------------------------------------------------------------------
// DCRNN on MI455X (gfx1250, wave32, WMMA).
//   conv0 (CIN=3): scalar (memory bound)
//   conv1 (H=128): edge scatter-add (f32 atomics) + WMMA GEMM [N,256]x[256,128]
//   LSTM x2: per-step fused WMMA kernel [1000,256]x[256,512] + gates in-regs
//     - A tile staged in LDS as f16 (convert once, ds_load_b128 fragments)
//     - B weights pre-packed fragment-major (global_load_b128 fragments)
//   pool + head + log_softmax: scalar
// ---------------------------------------------------------------------------

#define NN   100000
#define EE   1600000
#define HH   128
#define BG   100        // graphs == LSTM time steps
#define BN   1000       // nodes/graph == LSTM batch
#define SL   (BN * HH)  // one sequence slice = 128000 floats
#define APITCH 264      // LDS row pitch in halfs (132 dwords -> conflict-free)

typedef __attribute__((ext_vector_type(16))) _Float16 v16h;
typedef __attribute__((ext_vector_type(8)))  _Float16 v8h;
typedef __attribute__((ext_vector_type(8)))  float    v8f;

__device__ __forceinline__ v8f wmma_f16(v16h a, v16h b, v8f c) {
  return __builtin_amdgcn_wmma_f32_16x16x32_f16(false, a, false, b,
                                                (short)0, c, false, false);
}
__device__ __forceinline__ float sigmoidf(float v) {
  return 1.0f / (1.0f + __expf(-v));
}
__device__ __forceinline__ v16h combine16(v8h lo, v8h hi) {
  v16h r;
#pragma unroll
  for (int i = 0; i < 8; ++i) { r[i] = lo[i]; r[i + 8] = hi[i]; }
  return r;
}

// ----------------------------- weight prep --------------------------------
// Fragment-major packing. For column tile nt, K-chunk kk, lane, elem i:
//   Wp[((nt*8+kk)*32+lane)*16+i] = Wlogical[kk*32 + (i&7) + 8*(lane>>4) + 16*(i>>3)][nt*16 + (lane&15)]
// so a lane's v16h B fragment is 32 contiguous bytes (2x global_load_b128).

__global__ void k_prep_lstm(const float* __restrict__ Wih,
                            const float* __restrict__ Whh,
                            const float* __restrict__ bih,
                            const float* __restrict__ bhh,
                            _Float16* __restrict__ Wp, float* __restrict__ bc) {
  int idx = blockIdx.x * blockDim.x + threadIdx.x;      // 32*8*32*16 = 131072
  if (idx < 32 * 8 * 32 * 16) {
    int i = idx & 15, lane = (idx >> 4) & 31, kk = (idx >> 9) & 7, nt = idx >> 12;
    int j  = nt * 16 + (lane & 15);                     // output col (0..511)
    int kb = kk * 32 + (i & 7) + 8 * (lane >> 4) + 16 * (i >> 3);  // K index (0..255)
    float w = (kb < 128) ? Wih[j * 128 + kb] : Whh[j * 128 + (kb - 128)];
    Wp[idx] = (_Float16)w;
  }
  if (idx < 512) bc[idx] = bih[idx] + bhh[idx];
}

__global__ void k_prep_conv2(const float* __restrict__ Ws1,
                             const float* __restrict__ Wn1,
                             _Float16* __restrict__ Wp) {
  int idx = blockIdx.x * blockDim.x + threadIdx.x;      // 8*8*32*16 = 32768
  if (idx >= 8 * 8 * 32 * 16) return;
  int i = idx & 15, lane = (idx >> 4) & 31, kk = (idx >> 9) & 7, nt = idx >> 12;
  int j  = nt * 16 + (lane & 15);                       // output col (0..127)
  int kb = kk * 32 + (i & 7) + 8 * (lane >> 4) + 16 * (i >> 3);  // K index (0..255)
  float w = (kb < 128) ? Ws1[kb * 128 + j] : Wn1[(kb - 128) * 128 + j];
  Wp[idx] = (_Float16)w;
}

// ----------------------------- graph phase --------------------------------
__global__ void k_scatter0(const int* __restrict__ src, const int* __restrict__ dst,
                           const float* __restrict__ x,
                           float* __restrict__ agg0, float* __restrict__ deg) {
  int e = blockIdx.x * blockDim.x + threadIdx.x;
  if (e >= EE) return;
  int s = src[e], d = dst[e];
  atomicAdd(&deg[d], 1.0f);
  atomicAdd(&agg0[d * 3 + 0], x[s * 3 + 0]);
  atomicAdd(&agg0[d * 3 + 1], x[s * 3 + 1]);
  atomicAdd(&agg0[d * 3 + 2], x[s * 3 + 2]);
}

__global__ void k_conv0(const float* __restrict__ x, const float* __restrict__ agg0,
                        const float* __restrict__ deg,
                        const float* __restrict__ Ws, const float* __restrict__ Wn,
                        const float* __restrict__ b0, float* __restrict__ h0) {
  int idx = blockIdx.x * blockDim.x + threadIdx.x;      // N*128
  if (idx >= NN * HH) return;
  int n = idx >> 7, j = idx & 127;
  float dinv = 1.0f / fmaxf(deg[n], 1.0f);
  float acc = b0[j];
#pragma unroll
  for (int c = 0; c < 3; ++c)
    acc += x[n * 3 + c] * Ws[c * 128 + j] + (agg0[n * 3 + c] * dinv) * Wn[c * 128 + j];
  h0[idx] = fmaxf(acc, 0.0f);
}

__global__ void k_scatter1(const int* __restrict__ src, const int* __restrict__ dst,
                           const float* __restrict__ h0, float* __restrict__ agg) {
  int idx = blockIdx.x * blockDim.x + threadIdx.x;      // E*128
  int e = idx >> 7, j = idx & 127;
  if (e >= EE) return;
  atomicAdd(&agg[dst[e] * 128 + j], h0[src[e] * 128 + j]);
}

__global__ void k_normagg(const float* __restrict__ deg, float* __restrict__ agg) {
  int idx = blockIdx.x * blockDim.x + threadIdx.x;      // N*128
  if (idx >= NN * HH) return;
  agg[idx] *= 1.0f / fmaxf(deg[idx >> 7], 1.0f);
}

// conv1 transform: h1 = relu([h0 | aggn] @ W + b1). 16-row tile per block,
// 8 waves = 8 column tiles, K=256 staged once through LDS as f16.
__global__ void k_conv2(const float* __restrict__ h0, const float* __restrict__ aggn,
                        const _Float16* __restrict__ Wp, const float* __restrict__ b1,
                        float* __restrict__ h1) {
  __shared__ _Float16 As[16][APITCH];
  const int lane = threadIdx.x & 31;
  const int wave = threadIdx.x >> 5;
  const int grp  = lane >> 4, lcol = lane & 15;
  const int mt   = blockIdx.x;                          // 6250 (N % 16 == 0)
  // ---- stage A tile: thread -> (row r, 16-col chunk c) ----
  {
    const int r = threadIdx.x >> 4, c = threadIdx.x & 15;
    const size_t grow = (size_t)(mt * 16 + r);
    const float* sp = (c < 8) ? (h0 + grow * 128 + c * 16)
                              : (aggn + grow * 128 + (c - 8) * 16);
    v8h p0, p1;
#pragma unroll
    for (int i = 0; i < 8; ++i) {
      p0[i] = (_Float16)sp[i];
      p1[i] = (_Float16)sp[i + 8];
    }
    *(v8h*)&As[r][c * 16]     = p0;
    *(v8h*)&As[r][c * 16 + 8] = p1;
  }
  __syncthreads();

  v8f acc = {};
  const _Float16* w0 = Wp + ((size_t)(wave * 8) * 32 + lane) * 16;  // kk stride 512
#pragma unroll
  for (int kk = 0; kk < 8; ++kk) {
    const int o0 = kk * 32 + 8 * grp;
    v16h a = combine16(*(const v8h*)&As[lcol][o0],
                       *(const v8h*)&As[lcol][o0 + 16]);
    v16h b = *(const v16h*)(w0 + kk * 512);
    acc = wmma_f16(a, b, acc);
  }
#pragma unroll
  for (int v = 0; v < 8; ++v) {
    int m = mt * 16 + v + 8 * grp;
    int col = wave * 16 + lcol;
    h1[m * 128 + col] = fmaxf(acc[v] + b1[col], 0.0f);
  }
}

// ----------------------------- LSTM step ----------------------------------
// g = [xt|hprev] @ W + bc ; gates ; c in place ; h -> hnext (next seq slice).
// Wave = one 16-col hidden tile, 4 gate accumulators, 32 WMMAs per wave.
__global__ void k_lstm_step(const float* __restrict__ xt,
                            const float* __restrict__ hprev,
                            float* __restrict__ hnext,
                            float* __restrict__ cst,
                            const _Float16* __restrict__ Wp,   // packed [32][8][32][16]
                            const float* __restrict__ bc) {    // [512]
  __shared__ _Float16 As[16][APITCH];
  const int lane = threadIdx.x & 31;
  const int wave = threadIdx.x >> 5;
  const int grp  = lane >> 4, lcol = lane & 15;
  const int mt   = blockIdx.x;                          // 63 tiles over 1000 rows
  // ---- stage A tile ([xt | hprev], f32 -> f16) ----
  {
    const int r = threadIdx.x >> 4, c = threadIdx.x & 15;
    const int grow = mt * 16 + r;
    const bool valid = grow < BN;
    const float* sp = (c < 8) ? (xt + (size_t)grow * 128 + c * 16)
                              : (hprev + (size_t)grow * 128 + (c - 8) * 16);
    v8h p0, p1;
#pragma unroll
    for (int i = 0; i < 8; ++i) {
      p0[i] = (_Float16)(valid ? sp[i]     : 0.0f);
      p1[i] = (_Float16)(valid ? sp[i + 8] : 0.0f);
    }
    *(v8h*)&As[r][c * 16]     = p0;
    *(v8h*)&As[r][c * 16 + 8] = p1;
  }
  __syncthreads();

  v8f aI = {}, aF = {}, aG = {}, aO = {};
  // gate g, chunk kk fragment: w0 + g*32768 + kk*512 (halfs)
  const _Float16* w0 = Wp + ((size_t)(wave * 8) * 32 + lane) * 16;
#pragma unroll
  for (int kk = 0; kk < 8; ++kk) {
    const int o0 = kk * 32 + 8 * grp;
    v16h a = combine16(*(const v8h*)&As[lcol][o0],
                       *(const v8h*)&As[lcol][o0 + 16]);
    v16h bI = *(const v16h*)(w0 + kk * 512);
    v16h bF = *(const v16h*)(w0 + 32768 + kk * 512);
    v16h bG = *(const v16h*)(w0 + 65536 + kk * 512);
    v16h bO = *(const v16h*)(w0 + 98304 + kk * 512);
    aI = wmma_f16(a, bI, aI);
    aF = wmma_f16(a, bF, aF);
    aG = wmma_f16(a, bG, aG);
    aO = wmma_f16(a, bO, aO);
  }
#pragma unroll
  for (int v = 0; v < 8; ++v) {
    int m = mt * 16 + v + 8 * grp;
    int col = wave * 16 + lcol;
    if (m < BN) {
      float gi = aI[v] + bc[col];
      float gf = aF[v] + bc[col + 128];
      float gg = aG[v] + bc[col + 256];
      float go = aO[v] + bc[col + 384];
      float cn = sigmoidf(gf) * cst[m * 128 + col] + sigmoidf(gi) * tanhf(gg);
      cst[m * 128 + col] = cn;
      hnext[m * 128 + col] = sigmoidf(go) * tanhf(cn);
    }
  }
}

// ----------------------------- pool + head --------------------------------
__global__ void k_pool(const float* __restrict__ seq, float* __restrict__ pooled) {
  int b = blockIdx.x, j = threadIdx.x;        // 100 blocks x 128 threads
  float s = 0.0f;
  const float* base = seq + (size_t)b * SL + j;
  for (int n = 0; n < BN; ++n) s += base[n * 128];
  pooled[b * 128 + j] = s * (1.0f / (float)BN);
}

__global__ void k_head(const float* __restrict__ pooled,
                       const float* __restrict__ Wout,
                       const float* __restrict__ bout, float* __restrict__ out) {
  int b = threadIdx.x;
  if (b >= BG) return;
  float l0 = bout[0], l1 = bout[1];
  for (int k = 0; k < 128; ++k) {
    float p = pooled[b * 128 + k];
    l0 += p * Wout[k * 2 + 0];
    l1 += p * Wout[k * 2 + 1];
  }
  float mx  = fmaxf(l0, l1);
  float lse = mx + logf(__expf(l0 - mx) + __expf(l1 - mx));
  out[b * 2 + 0] = l0 - lse;
  out[b * 2 + 1] = l1 - lse;
}

// ----------------------------- driver -------------------------------------
extern "C" void kernel_launch(void* const* d_in, const int* in_sizes, int n_in,
                              void* d_out, int out_size, void* d_ws, size_t ws_size,
                              hipStream_t stream) {
  (void)in_sizes; (void)n_in; (void)out_size; (void)ws_size;
  const float* x    = (const float*)d_in[0];
  const int*   ei   = (const int*)d_in[1];
  const int*   src  = ei;
  const int*   dst  = ei + EE;
  const float* Ws0  = (const float*)d_in[3];
  const float* Wn0  = (const float*)d_in[4];
  const float* b0   = (const float*)d_in[5];
  const float* Ws1  = (const float*)d_in[6];
  const float* Wn1  = (const float*)d_in[7];
  const float* b1   = (const float*)d_in[8];
  const float* Wih0 = (const float*)d_in[9];
  const float* Whh0 = (const float*)d_in[10];
  const float* bih0 = (const float*)d_in[11];
  const float* bhh0 = (const float*)d_in[12];
  const float* Wih1 = (const float*)d_in[13];
  const float* Whh1 = (const float*)d_in[14];
  const float* bih1 = (const float*)d_in[15];
  const float* bhh1 = (const float*)d_in[16];
  const float* Wout = (const float*)d_in[17];
  const float* bout = (const float*)d_in[18];
  float* out = (float*)d_out;

  char* base = (char*)d_ws;
  size_t off = 0;
  auto take = [&](size_t bytes) -> char* {
    off = (off + 255) & ~(size_t)255;
    char* p = base + off;
    off += bytes;
    return p;
  };
  float*    deg   = (float*)take((size_t)NN * 4);
  float*    agg0  = (float*)take((size_t)NN * 3 * 4);
  float*    agg   = (float*)take((size_t)NN * HH * 4);
  float*    h0    = (float*)take((size_t)NN * HH * 4);   // conv0 out, later seq1
  float*    h1    = (float*)take((size_t)NN * HH * 4);   // conv1 out, later seq2
  float*    hzero = (float*)take((size_t)BN * HH * 4);
  float*    cbuf  = (float*)take((size_t)BN * HH * 4);
  _Float16* WpL0  = (_Float16*)take((size_t)32 * 8 * 32 * 16 * 2);
  _Float16* WpL1  = (_Float16*)take((size_t)32 * 8 * 32 * 16 * 2);
  _Float16* WpC2  = (_Float16*)take((size_t)8 * 8 * 32 * 16 * 2);
  float*    bcL0  = (float*)take(512 * 4);
  float*    bcL1  = (float*)take(512 * 4);
  float*    pooled= (float*)take((size_t)BG * HH * 4);

  // zero accumulators / initial states (graph-capture legal async memsets)
  hipMemsetAsync(deg,   0, (size_t)NN * 4,      stream);
  hipMemsetAsync(agg0,  0, (size_t)NN * 3 * 4,  stream);
  hipMemsetAsync(agg,   0, (size_t)NN * HH * 4, stream);
  hipMemsetAsync(hzero, 0, (size_t)BN * HH * 4, stream);
  hipMemsetAsync(cbuf,  0, (size_t)BN * HH * 4, stream);

  // weight prep (fragment-major f16 packing + bias fusion)
  k_prep_lstm<<<512, 256, 0, stream>>>(Wih0, Whh0, bih0, bhh0, WpL0, bcL0);
  k_prep_lstm<<<512, 256, 0, stream>>>(Wih1, Whh1, bih1, bhh1, WpL1, bcL1);
  k_prep_conv2<<<64, 256, 0, stream>>>(Ws1, Wn1, WpC2);

  // diffusion conv 0
  k_scatter0<<<(EE + 255) / 256, 256, 0, stream>>>(src, dst, x, agg0, deg);
  k_conv0<<<(NN * HH) / 256, 256, 0, stream>>>(x, agg0, deg, Ws0, Wn0, b0, h0);

  // diffusion conv 1 (scatter + normalize + WMMA transform)
  k_scatter1<<<(EE * HH) / 256, 256, 0, stream>>>(src, dst, h0, agg);
  k_normagg<<<(NN * HH) / 256, 256, 0, stream>>>(deg, agg);
  k_conv2<<<NN / 16, 256, 0, stream>>>(h0, agg, WpC2, b1, h1);

  // LSTM layer 1: input seq = h1, output seq1 -> h0 region
  for (int t = 0; t < BG; ++t) {
    const float* xt = h1 + (size_t)t * SL;
    const float* hp = (t == 0) ? hzero : h0 + (size_t)(t - 1) * SL;
    float* hn = h0 + (size_t)t * SL;
    k_lstm_step<<<(BN + 15) / 16, 256, 0, stream>>>(xt, hp, hn, cbuf, WpL0, bcL0);
  }

  // LSTM layer 2: input seq1 = h0, output seq2 -> h1 region
  hipMemsetAsync(cbuf, 0, (size_t)BN * HH * 4, stream);
  for (int t = 0; t < BG; ++t) {
    const float* xt = h0 + (size_t)t * SL;
    const float* hp = (t == 0) ? hzero : h1 + (size_t)(t - 1) * SL;
    float* hn = h1 + (size_t)t * SL;
    k_lstm_step<<<(BN + 15) / 16, 256, 0, stream>>>(xt, hp, hn, cbuf, WpL1, bcL1);
  }

  // pool + head
  k_pool<<<BG, 128, 0, stream>>>(h1, pooled);
  k_head<<<1, 128, 0, stream>>>(pooled, Wout, bout, out);
}
